// LocalToPair_4063039062767
// MI455X (gfx1250) — matrix operand order repaired and checked by reading, not compiled, not run on hardware
//
#include <hip/hip_runtime.h>

// ---------------------------------------------------------------------------
// Problem constants (from reference): N=8192, K=32, D=256, P=128, F=320
// ---------------------------------------------------------------------------
#define NN   8192
#define KK   32
#define DD   256
#define PP   128
#define FF   320          // P + 64 + P
#define HID  256          // 2P
#define KT_F 10           // FF / 32  k-tiles for gate/hid GEMM
#define KT_H 8            // HID / 32 k-tiles for out GEMM
#define NT_F 16           // HID / 16 n-tiles for gate/hid GEMM
#define NT_O 8            // PP / 16  n-tiles for out GEMM

#define FSTRIDE 328       // LDS feature row stride (elems, mult of 8 -> 16B aligned rows)
#define HSTRIDE 264       // LDS hidden  row stride
#define FEAT_BYTES (KK * FSTRIDE * 2)   // 20992, 16B aligned
#define POOL2_BYTES 32768               // staging (2x16KB) overlapped with hid tile

typedef __attribute__((ext_vector_type(16))) __bf16 v16bf;
typedef __attribute__((ext_vector_type(8)))  float  v8f;

union Frag16 {
    v16bf v;
    uint4 q[2];
};

__device__ __forceinline__ unsigned short f32_to_bf16(float f) {
    unsigned int u = __builtin_bit_cast(unsigned int, f);
    unsigned int r = u + 0x7FFFu + ((u >> 16) & 1u);   // round-to-nearest-even
    return (unsigned short)(r >> 16);
}

__device__ __forceinline__ float gelu_tanh(float x) {
    const float c = 0.7978845608028654f;               // sqrt(2/pi)
    float t = c * (x + 0.044715f * x * x * x);
    return 0.5f * x * (1.0f + tanhf(t));
}

// Low 32 bits of a flat pointer into the LDS aperture == LDS byte offset
// (ISA 10.2).
__device__ __forceinline__ unsigned lds_off(const void* p) {
    return (unsigned)(unsigned long long)(uintptr_t)p;
}

// Async global -> LDS copy of 16 bytes per lane (GLOBAL_LOAD_ASYNC_TO_LDS_B128,
// tracked by ASYNCcnt).  Inline asm keeps it portable across toolchains.
__device__ __forceinline__ void async_copy_b128(unsigned lds_byte_off, const void* gptr) {
    asm volatile("global_load_async_to_lds_b128 %0, %1, off"
                 :: "v"(lds_byte_off), "v"(gptr)
                 : "memory");
}

__device__ __forceinline__ void wait_asynccnt0() {
    asm volatile("s_wait_asynccnt 0" ::: "memory");
}

// ---------------------------------------------------------------------------
// Kernel 1: per-row projections of `local`
//   OLg = gelu(local @ W_ol) [N,8], ORf = local @ W_or [N,8]
//   AL  = local @ W_al [N,128],     AR  = local @ W_ar [N,128]
// ---------------------------------------------------------------------------
__global__ __launch_bounds__(256) void proj_kernel(
    const float* __restrict__ local,
    const float* __restrict__ W_ol, const float* __restrict__ W_or,
    const float* __restrict__ W_al, const float* __restrict__ W_ar,
    float* __restrict__ OLg, float* __restrict__ ORf,
    float* __restrict__ AL,  float* __restrict__ AR)
{
    __shared__ float x[DD];
    const int n = blockIdx.x;
    const int t = threadIdx.x;
    x[t] = local[n * DD + t];
    __syncthreads();

    for (int o = t; o < 272; o += 256) {
        const float* W; int col, ncol;
        if (o < 8)        { W = W_ol; col = o;       ncol = 8;   }
        else if (o < 16)  { W = W_or; col = o - 8;   ncol = 8;   }
        else if (o < 144) { W = W_al; col = o - 16;  ncol = PP;  }
        else              { W = W_ar; col = o - 144; ncol = PP;  }
        float acc = 0.0f;
        #pragma unroll 8
        for (int d = 0; d < DD; ++d) acc = fmaf(x[d], W[d * ncol + col], acc);
        if (o < 8)        OLg[n * 8  + col] = gelu_tanh(acc);
        else if (o < 16)  ORf[n * 8  + col] = acc;
        else if (o < 144) AL [n * PP + col] = acc;
        else              AR [n * PP + col] = acc;
    }
}

// ---------------------------------------------------------------------------
// Kernel 2: repack W_gate / W_hid / W_out (f32, row-major [K][N]) into bf16
// WMMA-B fragment-major layout:
//   frag(nt, kt) is 512 bf16: lane*16 + e, where
//     N = nt*16 + (lane & 15),  K = kt*32 + (e<8 ? 0 : 16) + (lane>=16 ? 8 : 0) + (e & 7)
// ---------------------------------------------------------------------------
__global__ __launch_bounds__(256) void wconv_kernel(
    const float* __restrict__ Wg, const float* __restrict__ Wh,
    const float* __restrict__ Wo,
    unsigned short* __restrict__ WgB, unsigned short* __restrict__ WhB,
    unsigned short* __restrict__ WoB)
{
    int i = blockIdx.x * 256 + threadIdx.x;
    if (i >= 2 * NT_F * KT_F * 512 + NT_O * KT_H * 512) return;

    const float* src; unsigned short* dst; int idx, nKt, ncol;
    if (i < NT_F * KT_F * 512)            { src = Wg; dst = WgB; idx = i;                    nKt = KT_F; ncol = HID; }
    else if (i < 2 * NT_F * KT_F * 512)   { src = Wh; dst = WhB; idx = i - NT_F*KT_F*512;    nKt = KT_F; ncol = HID; }
    else                                  { src = Wo; dst = WoB; idx = i - 2*NT_F*KT_F*512;  nKt = KT_H; ncol = PP;  }

    int nt   = idx / (nKt * 512);
    int rem  = idx % (nKt * 512);
    int kt   = rem / 512;
    int q    = rem % 512;
    int lane = q / 16;
    int e    = q % 16;
    int Kx   = kt * 32 + ((e < 8) ? 0 : 16) + ((lane >= 16) ? 8 : 0) + (e & 7);
    int Nx   = nt * 16 + (lane & 15);
    dst[idx] = f32_to_bf16(src[Kx * ncol + Nx]);
}

// ---------------------------------------------------------------------------
// Kernel 3: fused per-row pipeline. One block = one n (32 neighbour rows).
//   async-stage pair + gathered AR tiles into LDS (ASYNCcnt path), build
//   32x320 bf16 feature tile, WMMA gate/hid GEMMs (K=320), gelu*mul,
//   LDS hidden tile (32x256 bf16, overlapping the staging pool), WMMA out GEMM.
// ---------------------------------------------------------------------------
__global__ __launch_bounds__(128) void fused_kernel(
    const float* __restrict__ pair,
    const long long* __restrict__ nbr,
    const float* __restrict__ OLg, const float* __restrict__ ORf,
    const float* __restrict__ AL,  const float* __restrict__ AR,
    const unsigned short* __restrict__ WgB,
    const unsigned short* __restrict__ WhB,
    const unsigned short* __restrict__ WoB,
    float* __restrict__ out)
{
    __shared__ __align__(16) char smem[FEAT_BYTES + POOL2_BYTES];
    unsigned short* feat = (unsigned short*)smem;
    float* stage_pair    = (float*)(smem + FEAT_BYTES);            // 16384 B
    float* stage_ar      = (float*)(smem + FEAT_BYTES + 16384);    // 16384 B
    unsigned short* hid  = (unsigned short*)(smem + FEAT_BYTES);   // reused after build

    __shared__ float s_ol[8];
    __shared__ float s_or[KK * 8];
    __shared__ int   s_nb[KK];

    const int n    = blockIdx.x;
    const int tid  = threadIdx.x;
    const int lane = tid & 31;
    const int wave = tid >> 5;          // 0..3

    // keep weight fragments warm in GL2 (global_prefetch_b8)
    __builtin_prefetch(WgB + (size_t)wave * (NT_F/4) * KT_F * 512, 0, 0);
    __builtin_prefetch(WhB + (size_t)wave * (NT_F/4) * KT_F * 512, 0, 0);

    if (tid < KK) s_nb[tid] = (int)nbr[(long long)n * KK + tid];
    if (tid < 8)  s_ol[tid] = OLg[n * 8 + tid];
    __syncthreads();

    // ---- async global->LDS staging: pair tile (HBM) + gathered AR rows (L2)
    #pragma unroll
    for (int j = 0; j < 8; ++j) {
        int i  = tid + j * 128;        // 0..1023 float4 index
        int r  = i >> 5;
        int c4 = (i & 31) << 2;
        const float* gp = pair + ((size_t)n * KK + r) * PP + c4;
        async_copy_b128(lds_off(&stage_pair[r * PP + c4]), gp);
        const float* ga = AR + (size_t)s_nb[r] * PP + c4;
        async_copy_b128(lds_off(&stage_ar[r * PP + c4]), ga);
    }

    // overlap async latency with the neighbour outer_right gather
    for (int i = tid; i < KK * 8; i += 128) {
        int k = i >> 3, j = i & 7;
        s_or[i] = ORf[s_nb[k] * 8 + j];
    }

    wait_asynccnt0();
    __syncthreads();

    // ---- build bf16 feature tile from staged f32 data
    for (int i = tid; i < KK * PP; i += 128) {
        int r = i >> 7, c = i & 127;
        feat[r * FSTRIDE + c] = f32_to_bf16(stage_pair[i]);            // pair
        float v = AL[n * PP + c] + stage_ar[i];                        // additive
        feat[r * FSTRIDE + 192 + c] = f32_to_bf16(v);
    }
    for (int i = tid; i < KK * 64; i += 128) {                         // outer 8x8
        int r = i >> 6, c = i & 63;
        feat[r * FSTRIDE + 128 + c] = f32_to_bf16(s_ol[c >> 3] * s_or[r * 8 + (c & 7)]);
    }
    __syncthreads();

    // ---- gate / hid GEMMs: M=32, K=320, N=256. Wave w owns N cols [w*64, w*64+64)
    const int hiSel = (lane >= 16) ? 1 : 0;
    const int arow  = lane & 15;

    for (int nt4 = 0; nt4 < 4; ++nt4) {
        const int nt = wave * 4 + nt4;
        for (int mt = 0; mt < 2; ++mt) {
            v8f accg = {};
            v8f acch = {};
            for (int kt = 0; kt < KT_F; ++kt) {
                // A fragment from LDS feature tile (16-bit A layout)
                Frag16 a;
                int M  = mt * 16 + arow;
                int kb = kt * 32 + (hiSel ? 8 : 0);
                a.q[0] = *(const uint4*)&feat[M * FSTRIDE + kb];
                a.q[1] = *(const uint4*)&feat[M * FSTRIDE + kb + 16];
                // B fragments from fragment-major bf16 weights (L2-resident)
                Frag16 bg, bh;
                const size_t fb = ((size_t)(nt * KT_F + kt) * 512) + (size_t)lane * 16;
                bg.q[0] = *(const uint4*)(WgB + fb);
                bg.q[1] = *(const uint4*)(WgB + fb + 8);
                bh.q[0] = *(const uint4*)(WhB + fb);
                bh.q[1] = *(const uint4*)(WhB + fb + 8);
                accg = __builtin_amdgcn_wmma_f32_16x16x32_bf16(
                    false, a.v, false, bg.v, (short)0, accg, false, false);
                acch = __builtin_amdgcn_wmma_f32_16x16x32_bf16(
                    false, a.v, false, bh.v, (short)0, acch, false, false);
            }
            // hidden = gelu(gate) * hid  -> LDS bf16
            #pragma unroll
            for (int r = 0; r < 8; ++r) {
                float hv = gelu_tanh(accg[r]) * acch[r];
                int M  = mt * 16 + r + (hiSel ? 8 : 0);
                int Nc = nt * 16 + arow;
                hid[M * HSTRIDE + Nc] = f32_to_bf16(hv);
            }
        }
    }
    __syncthreads();

    // ---- out GEMM: M=32, K=256, N=128. 16 tiles over 4 waves.
    for (int t4 = 0; t4 < 4; ++t4) {
        const int tile = wave * 4 + t4;
        const int nt = tile & 7;
        const int mt = tile >> 3;
        v8f acc = {};
        for (int kt = 0; kt < KT_H; ++kt) {
            Frag16 a, b;
            int M  = mt * 16 + arow;
            int kb = kt * 32 + (hiSel ? 8 : 0);
            a.q[0] = *(const uint4*)&hid[M * HSTRIDE + kb];
            a.q[1] = *(const uint4*)&hid[M * HSTRIDE + kb + 16];
            const size_t fb = ((size_t)(nt * KT_H + kt) * 512) + (size_t)lane * 16;
            b.q[0] = *(const uint4*)(WoB + fb);
            b.q[1] = *(const uint4*)(WoB + fb + 8);
            acc = __builtin_amdgcn_wmma_f32_16x16x32_bf16(
                false, a.v, false, b.v, (short)0, acc, false, false);
        }
        #pragma unroll
        for (int r = 0; r < 8; ++r) {
            int M  = mt * 16 + r + (hiSel ? 8 : 0);
            int Nc = nt * 16 + arow;
            out[((long long)n * KK + M) * PP + Nc] = acc[r];
        }
    }
}

// ---------------------------------------------------------------------------
// Host-side launcher
// ---------------------------------------------------------------------------
extern "C" void kernel_launch(void* const* d_in, const int* in_sizes, int n_in,
                              void* d_out, int out_size, void* d_ws, size_t ws_size,
                              hipStream_t stream) {
    const float*     local = (const float*)d_in[0];
    const float*     pair  = (const float*)d_in[1];
    const long long* nbr   = (const long long*)d_in[2];
    const float*     W_ol  = (const float*)d_in[3];
    const float*     W_or  = (const float*)d_in[4];
    const float*     W_al  = (const float*)d_in[5];
    const float*     W_ar  = (const float*)d_in[6];
    const float*     W_g   = (const float*)d_in[7];
    const float*     W_h   = (const float*)d_in[8];
    const float*     W_o   = (const float*)d_in[9];
    float* out = (float*)d_out;

    char* ws = (char*)d_ws;
    size_t off = 0;
    float* OLg = (float*)(ws + off); off += (size_t)NN * 8  * sizeof(float);
    float* ORf = (float*)(ws + off); off += (size_t)NN * 8  * sizeof(float);
    float* AL  = (float*)(ws + off); off += (size_t)NN * PP * sizeof(float);
    float* AR  = (float*)(ws + off); off += (size_t)NN * PP * sizeof(float);
    unsigned short* WgB = (unsigned short*)(ws + off); off += (size_t)NT_F * KT_F * 512 * 2;
    unsigned short* WhB = (unsigned short*)(ws + off); off += (size_t)NT_F * KT_F * 512 * 2;
    unsigned short* WoB = (unsigned short*)(ws + off); off += (size_t)NT_O * KT_H * 512 * 2;

    proj_kernel<<<NN, 256, 0, stream>>>(local, W_ol, W_or, W_al, W_ar,
                                        OLg, ORf, AL, AR);

    const int wtotal = 2 * NT_F * KT_F * 512 + NT_O * KT_H * 512;  // 196608
    wconv_kernel<<<(wtotal + 255) / 256, 256, 0, stream>>>(W_g, W_h, W_o,
                                                           WgB, WhB, WoB);

    fused_kernel<<<NN, 128, 0, stream>>>(pair, nbr, OLg, ORf, AL, AR,
                                         WgB, WhB, WoB, out);
}
